// GraphConv_v3_22067541967340
// MI455X (gfx1250) — matrix operand order repaired and checked by reading, not compiled, not run on hardware
//
#include <hip/hip_runtime.h>
#include <hip/hip_bf16.h>

// ---------------------------------------------------------------------------
// Types for CDNA5 WMMA (gfx1250, wave32)
// ---------------------------------------------------------------------------
typedef __attribute__((ext_vector_type(16))) __bf16 bf16x16;
typedef __attribute__((ext_vector_type(8)))  __bf16 bf16x8;
typedef __attribute__((ext_vector_type(8)))  float  v8f;

union FragU { bf16x16 v; bf16x8 h[2]; };

#define WMMA_BF16(a, b, c) \
  __builtin_amdgcn_wmma_f32_16x16x32_bf16(false, (a), false, (b), (short)0, (c), false, false)

static constexpr int D    = 128;
static constexpr int HID  = 256;

// ---------------------------------------------------------------------------
// Helpers
// ---------------------------------------------------------------------------
__device__ __forceinline__ unsigned short f2bfu(float f) {
  unsigned u = __float_as_uint(f);
  u += 0x7FFFu + ((u >> 16) & 1u);            // round-to-nearest-even
  return (unsigned short)(u >> 16);
}

__device__ __forceinline__ float sigmoidf_fast(float t) {
  return 1.0f / (1.0f + __expf(-t));
}
__device__ __forceinline__ float tanhf_fast(float t) {
  return 2.0f * sigmoidf_fast(2.0f * t) - 1.0f;
}

// A-fragment: 16x32 bf16 tile, row-major source (ISA 7.12.2 16-bit A layout).
// lane L (half = L>>4, m = L&15): elems 0..7 = K[k0+half*8 .. +8),
//                                 elems 8..15 = K[k0+16+half*8 .. +8)
__device__ __forceinline__ bf16x16 load_fragA(const __bf16* base, int ld,
                                              int row0, int k0, int lane) {
  int half = lane >> 4;
  int m    = lane & 15;
  const __bf16* p = base + (size_t)(row0 + m) * ld + (k0 + half * 8);
  FragU f;
  f.h[0] = *(const bf16x8*)(p);
  f.h[1] = *(const bf16x8*)(p + 16);
  return f.v;
}

// B-fragment: B = W^T (K x 16). Column n of B = row (col0+n) of row-major W.
// lane half h: elems 0..7 = K[k0+h*8 .. +8), elems 8..15 = K[k0+16+h*8 .. +8) (matches A).
__device__ __forceinline__ bf16x16 load_fragB(const __bf16* base, int ld,
                                              int col0, int k0, int lane) {
  int half = lane >> 4;
  int n    = lane & 15;
  const __bf16* p = base + (size_t)(col0 + n) * ld + (k0 + half * 8);
  FragU f;
  f.h[0] = *(const bf16x8*)(p);
  f.h[1] = *(const bf16x8*)(p + 16);
  return f.v;
}

__device__ __forceinline__ void atomicMaxFloat(float* addr, float v) {
  // Sign-aware int trick: single hardware atomic instead of a CAS loop.
  if (v >= 0.0f) atomicMax((int*)addr, __float_as_int(v));
  else           atomicMin((unsigned int*)addr, __float_as_uint(v));
}

// ---------------------------------------------------------------------------
// Elementwise kernels
// ---------------------------------------------------------------------------
__global__ void init_kernel(const float* __restrict__ x,
                            unsigned short* __restrict__ x_bf,
                            float* __restrict__ aggsum,
                            float* __restrict__ aggmax,
                            float* __restrict__ stats, int nd) {
  int i = blockIdx.x * blockDim.x + threadIdx.x;
  if (i < 256) stats[i] = 0.0f;
  if (i >= nd) return;
  x_bf[i]   = f2bfu(x[i]);
  aggsum[i] = 0.0f;
  aggmax[i] = __uint_as_float(0xFF800000u);   // -inf
}

__global__ void cvt_bf16_kernel(const float* __restrict__ s,
                                unsigned short* __restrict__ d, int n) {
  int i = blockIdx.x * blockDim.x + threadIdx.x;
  if (i < n) d[i] = f2bfu(s[i]);
}

// Each thread handles one float4 chunk (4 features) of one edge.
__global__ void scatter_kernel(const int* __restrict__ edges, int E,
                               const float* __restrict__ x,
                               const float* __restrict__ xi,
                               float* __restrict__ aggsum,
                               float* __restrict__ aggmax) {
  int tid = blockIdx.x * blockDim.x + threadIdx.x;
  int e = tid >> 5;
  if (e >= E) return;
  int f = (tid & 31) * 4;
  int src = edges[e];
  int dst = edges[E + e];
  float4 xs  = *(const float4*)(x  + (size_t)src * D + f);
  float4 xis = *(const float4*)(xi + (size_t)src * D + f);
  float* as = aggsum + (size_t)dst * D + f;
  float* am = aggmax + (size_t)dst * D + f;
  atomicAdd(as + 0, xs.x); atomicAdd(as + 1, xs.y);
  atomicAdd(as + 2, xs.z); atomicAdd(as + 3, xs.w);
  atomicMaxFloat(am + 0, xis.x); atomicMaxFloat(am + 1, xis.y);
  atomicMaxFloat(am + 2, xis.z); atomicMaxFloat(am + 3, xis.w);
}

// agg_max: -inf -> 0, convert to bf16; pack agg_sum (bf16) into cat[:, 0:D].
__global__ void sanitize_kernel(const float* __restrict__ aggmax,
                                const float* __restrict__ aggsum,
                                unsigned short* __restrict__ aggmax_bf,
                                unsigned short* __restrict__ cat_bf, int nd) {
  int i = blockIdx.x * blockDim.x + threadIdx.x;
  if (i >= nd) return;
  int row = i >> 7;       // / D
  int col = i & (D - 1);
  float m = aggmax[i];
  if (!(m >= -1e37f)) m = 0.0f;               // empty segments
  aggmax_bf[i] = f2bfu(m);
  cat_bf[(size_t)row * (2 * D) + col] = f2bfu(aggsum[i]);
}

// ---------------------------------------------------------------------------
// Generic WMMA bf16 GEMM: out(N x M) = A(N x K) @ W(M x K)^T, fused epilogues.
// One wave -> 16 rows x 64 cols (4 tiles, A-fragment reused 4x per K-step).
// EPI: 0 = f32 (+bias)          -> outF      (xi)
//      1 = bf16 (+bias +eps*X)  -> outB      (merge)
//      2 = bf16 relu(+bias)     -> outB      (backbone 1)
//      3 = f32  relu(+bias)     -> outF      (backbone 2)
// ---------------------------------------------------------------------------
template <int EPI>
__global__ void __launch_bounds__(128)
gemm_bf16_kernel(const __bf16* __restrict__ A, int lda,
                 const __bf16* __restrict__ W, int ldw,
                 const float* __restrict__ bias,
                 float* __restrict__ outF, unsigned short* __restrict__ outB, int ldo,
                 const float* __restrict__ X, const float* __restrict__ epsPtr,
                 int N, int K, int M) {
  int lane   = threadIdx.x;
  int waveId = blockIdx.x * blockDim.y + threadIdx.y;
  int nrt    = N >> 4;
  int ncg    = M >> 6;
  if (waveId >= nrt * ncg) return;            // wave-uniform: EXEC stays all-1s
  int rt = waveId % nrt;
  int cg = waveId / nrt;
  int row0 = rt << 4;
  int col0 = cg << 6;

  v8f acc[4] = {};
  for (int k = 0; k < K; k += 32) {
    bf16x16 a = load_fragA(A, lda, row0, k, lane);
#pragma unroll
    for (int t = 0; t < 4; ++t) {
      bf16x16 b = load_fragB(W, ldw, col0 + t * 16, k, lane);
      acc[t] = WMMA_BF16(a, b, acc[t]);
    }
  }

  int half = lane >> 4;
  int nl   = lane & 15;
  float epsv = (EPI == 1) ? *epsPtr : 0.0f;
#pragma unroll
  for (int t = 0; t < 4; ++t) {
    int n = col0 + t * 16 + nl;
    float bv = bias[n];
#pragma unroll
    for (int r = 0; r < 8; ++r) {
      int row = row0 + r + 8 * half;
      float v = acc[t][r] + bv;
      if (EPI == 1) v += epsv * X[(size_t)row * M + n];
      if (EPI == 2 || EPI == 3) v = fmaxf(v, 0.0f);
      if (EPI == 0 || EPI == 3) outF[(size_t)row * ldo + n] = v;
      else                      outB[(size_t)row * ldo + n] = f2bfu(v);
    }
  }
}

// ---------------------------------------------------------------------------
// Fused GRU: per wave, 6 WMMA accumulators (ir,iz,in | hr,hz,hn) over K=128,
// then the gate math; writes rnn_out (bf16) into cat[:, D:2D].
// ---------------------------------------------------------------------------
__global__ void __launch_bounds__(128)
gru_kernel(const __bf16* __restrict__ aggmax_bf,
           const __bf16* __restrict__ x_bf,
           const __bf16* __restrict__ Wih,   // (3D x D) row-major, bf16
           const __bf16* __restrict__ Whh,   // (3D x D)
           const float* __restrict__ b_ih, const float* __restrict__ b_hh,
           const float* __restrict__ x,      // f32, for the z-blend
           unsigned short* __restrict__ cat_bf, int N) {
  int lane   = threadIdx.x;
  int waveId = blockIdx.x * blockDim.y + threadIdx.y;
  int nrt    = N >> 4;
  if (waveId >= nrt * (D / 16)) return;
  int rt = waveId % nrt;
  int ct = waveId / nrt;
  int row0 = rt << 4;
  int col0 = ct << 4;

  v8f air = {}, aiz = {}, ain = {}, ahr = {}, ahz = {}, ahn = {};
#pragma unroll
  for (int k = 0; k < D; k += 32) {
    bf16x16 a1 = load_fragA(aggmax_bf, D, row0, k, lane);
    bf16x16 a2 = load_fragA(x_bf,      D, row0, k, lane);
    bf16x16 b;
    b = load_fragB(Wih, D, col0,         k, lane); air = WMMA_BF16(a1, b, air);
    b = load_fragB(Wih, D, D + col0,     k, lane); aiz = WMMA_BF16(a1, b, aiz);
    b = load_fragB(Wih, D, 2 * D + col0, k, lane); ain = WMMA_BF16(a1, b, ain);
    b = load_fragB(Whh, D, col0,         k, lane); ahr = WMMA_BF16(a2, b, ahr);
    b = load_fragB(Whh, D, D + col0,     k, lane); ahz = WMMA_BF16(a2, b, ahz);
    b = load_fragB(Whh, D, 2 * D + col0, k, lane); ahn = WMMA_BF16(a2, b, ahn);
  }

  int half = lane >> 4;
  int n    = col0 + (lane & 15);
  float bir = b_ih[n], biz = b_ih[D + n], bin = b_ih[2 * D + n];
  float bhr = b_hh[n], bhz = b_hh[D + n], bhn = b_hh[2 * D + n];
#pragma unroll
  for (int r = 0; r < 8; ++r) {
    int row = row0 + r + 8 * half;
    float xv = x[(size_t)row * D + n];
    float rg = sigmoidf_fast(air[r] + bir + ahr[r] + bhr);
    float zg = sigmoidf_fast(aiz[r] + biz + ahz[r] + bhz);
    float nc = tanhf_fast(ain[r] + bin + rg * (ahn[r] + bhn));
    float o  = (1.0f - zg) * nc + zg * xv;
    cat_bf[(size_t)row * (2 * D) + D + n] = f2bfu(o);
  }
}

// ---------------------------------------------------------------------------
// BatchNorm: column sums via LDS reduce + atomics, then normalize.
// ---------------------------------------------------------------------------
__global__ void bn_stats_kernel(const float* __restrict__ h2,
                                float* __restrict__ stats, int N) {
  __shared__ float ss[256];
  __shared__ float ss2[256];
  int tid = threadIdx.x;
  int c   = tid & (D - 1);
  int r0  = blockIdx.x * 256 + (tid >> 7);
  int rend = min(blockIdx.x * 256 + 256, N);
  float s = 0.0f, s2 = 0.0f;
  for (int row = r0; row < rend; row += 2) {
    float v = h2[(size_t)row * D + c];
    s += v; s2 += v * v;
  }
  ss[tid] = s; ss2[tid] = s2;
  __syncthreads();
  if (tid < 128) {
    atomicAdd(&stats[c],       ss[tid]  + ss[tid + 128]);
    atomicAdd(&stats[D + c],   ss2[tid] + ss2[tid + 128]);
  }
}

__global__ void bn_final_kernel(const float* __restrict__ h2,
                                const float* __restrict__ stats,
                                const float* __restrict__ gamma,
                                const float* __restrict__ beta,
                                float* __restrict__ out, float invN, int nd) {
  int i = blockIdx.x * blockDim.x + threadIdx.x;
  if (i >= nd) return;
  int c = i & (D - 1);
  float mean = stats[c] * invN;
  float var  = stats[D + c] * invN - mean * mean;
  out[i] = gamma[c] * (h2[i] - mean) * rsqrtf(var + 1e-5f) + beta[c];
}

// ---------------------------------------------------------------------------
// Host launcher
// ---------------------------------------------------------------------------
extern "C" void kernel_launch(void* const* d_in, const int* in_sizes, int n_in,
                              void* d_out, int out_size, void* d_ws, size_t ws_size,
                              hipStream_t stream) {
  const float* x       = (const float*)d_in[0];
  const int*   edges   = (const int*)d_in[1];
  const float* W_aff   = (const float*)d_in[2];
  const float* b_aff   = (const float*)d_in[3];
  const float* W_ih    = (const float*)d_in[4];
  const float* b_ih    = (const float*)d_in[5];
  const float* W_hh    = (const float*)d_in[6];
  const float* b_hh    = (const float*)d_in[7];
  const float* W_merge = (const float*)d_in[8];
  const float* b_merge = (const float*)d_in[9];
  const float* epsP    = (const float*)d_in[10];
  const float* W1      = (const float*)d_in[11];
  const float* b1      = (const float*)d_in[12];
  const float* W2      = (const float*)d_in[13];
  const float* b2      = (const float*)d_in[14];
  const float* gamma   = (const float*)d_in[15];
  const float* beta    = (const float*)d_in[16];

  const int N  = in_sizes[0] / D;
  const int E  = in_sizes[1] / 2;
  const int ND = N * D;

  // --- workspace carve-out (256B aligned) ---
  char* ws = (char*)d_ws;
  size_t o = 0;
  auto carve = [&](size_t bytes) -> void* {
    o = (o + 255) & ~(size_t)255;
    void* p = ws + o;
    o += bytes;
    return p;
  };
  unsigned short* x_bf      = (unsigned short*)carve((size_t)ND * 2);
  float*          aggsum    = (float*)carve((size_t)ND * 4);
  float*          aggmax    = (float*)carve((size_t)ND * 4);
  float*          xi        = (float*)carve((size_t)ND * 4);
  unsigned short* aggmax_bf = (unsigned short*)carve((size_t)ND * 2);
  unsigned short* cat_bf    = (unsigned short*)carve((size_t)N * 2 * D * 2);
  unsigned short* Waff_bf   = (unsigned short*)carve(D * D * 2);
  unsigned short* Wih_bf    = (unsigned short*)carve(3 * D * D * 2);
  unsigned short* Whh_bf    = (unsigned short*)carve(3 * D * D * 2);
  unsigned short* Wmrg_bf   = (unsigned short*)carve(D * 2 * D * 2);
  unsigned short* W1_bf     = (unsigned short*)carve(HID * D * 2);
  unsigned short* W2_bf     = (unsigned short*)carve(D * HID * 2);
  float*          stats     = (float*)carve(256 * 4);
  // Phase-disjoint aliases (lifetimes do not overlap):
  unsigned short* h_bf  = (unsigned short*)aggmax;   // N x D   bf16
  unsigned short* h1_bf = (unsigned short*)aggsum;   // N x HID bf16
  float*          h2    = xi;                        // N x D   f32

  const dim3 gb(32, 4);

  // 1) init + weight converts
  init_kernel<<<(ND + 255) / 256, 256, 0, stream>>>(x, x_bf, aggsum, aggmax, stats, ND);
  cvt_bf16_kernel<<<(D * D + 255) / 256, 256, 0, stream>>>(W_aff, Waff_bf, D * D);
  cvt_bf16_kernel<<<(3 * D * D + 255) / 256, 256, 0, stream>>>(W_ih, Wih_bf, 3 * D * D);
  cvt_bf16_kernel<<<(3 * D * D + 255) / 256, 256, 0, stream>>>(W_hh, Whh_bf, 3 * D * D);
  cvt_bf16_kernel<<<(D * 2 * D + 255) / 256, 256, 0, stream>>>(W_merge, Wmrg_bf, D * 2 * D);
  cvt_bf16_kernel<<<(HID * D + 255) / 256, 256, 0, stream>>>(W1, W1_bf, HID * D);
  cvt_bf16_kernel<<<(D * HID + 255) / 256, 256, 0, stream>>>(W2, W2_bf, D * HID);

  // 2) xi = x @ W_aff^T + b_aff   (f32 out, needed for scatter-max)
  {
    int waves = (N >> 4) * (D >> 6);
    gemm_bf16_kernel<0><<<(waves + 3) / 4, gb, 0, stream>>>(
        (const __bf16*)x_bf, D, (const __bf16*)Waff_bf, D, b_aff,
        xi, nullptr, D, nullptr, nullptr, N, D, D);
  }

  // 3) edge scatter: atomic add (sum) + atomic float-max
  scatter_kernel<<<((size_t)E * 32 + 255) / 256, 256, 0, stream>>>(
      edges, E, x, xi, aggsum, aggmax);

  // 4) sanitize agg_max, pack agg_sum -> cat[:, 0:D]
  sanitize_kernel<<<(ND + 255) / 256, 256, 0, stream>>>(
      aggmax, aggsum, aggmax_bf, cat_bf, ND);

  // 5) fused GRU -> cat[:, D:2D]
  {
    int waves = (N >> 4) * (D / 16);
    gru_kernel<<<(waves + 3) / 4, gb, 0, stream>>>(
        (const __bf16*)aggmax_bf, (const __bf16*)x_bf,
        (const __bf16*)Wih_bf, (const __bf16*)Whh_bf,
        b_ih, b_hh, x, cat_bf, N);
  }

  // 6) h = cat @ W_merge^T + b_merge + eps*x  -> bf16
  {
    int waves = (N >> 4) * (D >> 6);
    gemm_bf16_kernel<1><<<(waves + 3) / 4, gb, 0, stream>>>(
        (const __bf16*)cat_bf, 2 * D, (const __bf16*)Wmrg_bf, 2 * D, b_merge,
        nullptr, h_bf, D, x, epsP, N, 2 * D, D);
  }

  // 7) h1 = relu(h @ W1^T + b1) -> bf16
  {
    int waves = (N >> 4) * (HID >> 6);
    gemm_bf16_kernel<2><<<(waves + 3) / 4, gb, 0, stream>>>(
        (const __bf16*)h_bf, D, (const __bf16*)W1_bf, D, b1,
        nullptr, h1_bf, HID, nullptr, nullptr, N, D, HID);
  }

  // 8) h2 = relu(h1 @ W2^T + b2) -> f32
  {
    int waves = (N >> 4) * (D >> 6);
    gemm_bf16_kernel<3><<<(waves + 3) / 4, gb, 0, stream>>>(
        (const __bf16*)h1_bf, HID, (const __bf16*)W2_bf, HID, b2,
        h2, nullptr, D, nullptr, nullptr, N, HID, D);
  }

  // 9) batchnorm (train-mode, biased variance)
  bn_stats_kernel<<<(N + 255) / 256, 256, 0, stream>>>(h2, stats, N);
  bn_final_kernel<<<(ND + 255) / 256, 256, 0, stream>>>(
      h2, stats, gamma, beta, (float*)d_out, 1.0f / (float)N, ND);
}
